// EViT_62904091017672
// MI455X (gfx1250) — matrix operand order repaired and checked by reading, not compiled, not run on hardware
//
#include <hip/hip_runtime.h>
#include <hip/hip_bf16.h>
#include <math.h>
#include <stdint.h>

// ---------------- problem constants ----------------
constexpr int Bq   = 4;
constexpr int Nseq = 1570;
constexpr int Cdim = 768;
constexpr int Hn   = 12;
constexpr int DHd  = 64;
constexpr int Mtok = Nseq - 2;          // 1568
constexpr int KSEL = 1098;              // ceil(0.7*1568)
constexpr int CKn  = Mtok - KSEL;       // 470
constexpr int N2   = 2 + KSEL + 1;      // 1101
constexpr int TOK1 = Bq * Nseq;         // 6280
constexpr int TOK2 = Bq * N2;           // 4404
constexpr int NPAD = 1600;              // padded key dim for V^T
constexpr float SCALE_F = 0.125f;       // 64^-0.5
constexpr float LN_EPS  = 1e-5f;
constexpr int BSTRIDE = 40;             // padded LDS stride (elems) per B column

// ---------------- types ----------------
typedef __bf16 bf16_t;
typedef __attribute__((ext_vector_type(16))) __bf16 bf16x16;
typedef __attribute__((ext_vector_type(8)))  __bf16 bf16x8;
typedef __attribute__((ext_vector_type(8)))  float  f32x8;

__device__ inline f32x8 wmma_bf16(bf16x16 a, bf16x16 b, f32x8 c) {
  return __builtin_amdgcn_wmma_f32_16x16x32_bf16(
      /*neg_a=*/false, a, /*neg_b=*/false, b,
      /*c_mod=*/(short)0, c, /*reuse_a=*/false, /*reuse_b=*/false);
}

// A fragment: 16x32 bf16, row-major source. lane<16: row=lane&15, K {0..15};
// lane>=16: row=lane&15, K {16..31}.
__device__ inline bf16x16 load_a_frag(const bf16_t* A, int lda, int row, int k0, int lane) {
  const bf16_t* p = A + (size_t)row * lda + k0 + ((lane & 16) ? 16 : 0);
  union { bf16x16 v; bf16x8 h[2]; } u;
  u.h[0] = *(const bf16x8*)(p);
  u.h[1] = *(const bf16x8*)(p + 8);
  return u.v;
}

// B fragment: 32x16 bf16, column-major source Bt[col][k]. lane holds column
// `col`, 16 consecutive K starting at k0 (+16 for lanes 16..31).
__device__ inline bf16x16 load_b_frag(const bf16_t* Bt, int ldb, int col, int k0, int lane) {
  const bf16_t* p = Bt + (size_t)col * ldb + k0 + ((lane & 16) ? 16 : 0);
  union { bf16x16 v; bf16x8 h[2]; } u;
  u.h[0] = *(const bf16x8*)(p);
  u.h[1] = *(const bf16x8*)(p + 8);
  return u.v;
}

// B fragment from an LDS-staged 64x32 slab (padded stride BSTRIDE)
__device__ inline bf16x16 load_b_frag_lds(const bf16_t* Bs, int col, int lane) {
  const bf16_t* p = Bs + col * BSTRIDE + ((lane & 16) ? 16 : 0);
  union { bf16x16 v; bf16x8 h[2]; } u;
  u.h[0] = *(const bf16x8*)(p);
  u.h[1] = *(const bf16x8*)(p + 8);
  return u.v;
}

// ---- CDNA5 async global->LDS copy (ASYNCcnt-tracked, no VGPR round trip) ----
__device__ inline void async_ld_b128(uint32_t lds_byte_addr, const void* gptr) {
  asm volatile("global_load_async_to_lds_b128 %0, %1, off"
               :: "v"(lds_byte_addr), "v"((uint64_t)(uintptr_t)gptr)
               : "memory");
}
__device__ inline void wait_async0() {
  asm volatile("s_wait_asynccnt 0x0" ::: "memory");
}
__device__ inline void wait_async1() {
  asm volatile("s_wait_asynccnt 0x1" ::: "memory");
}

// 256 threads cooperatively stage a 64-col x 32-k bf16 slab of Bt into LDS.
// Each thread issues one async 16B transfer.
__device__ inline void stage_b_async(const bf16_t* Bt, int Kdim, int col0, int k0,
                                     bf16_t* buf, int tid) {
  int col = tid >> 2;             // 0..63
  int ko  = (tid & 3) * 8;        // 0,8,16,24
  const bf16_t* src = Bt + (size_t)(col0 + col) * Kdim + k0 + ko;
  uint32_t dst = (uint32_t)(uintptr_t)(buf + col * BSTRIDE + ko);
  async_ld_b128(dst, src);
}

// ---------------------------------------------------------------------
// Shared GEMM main loop: 8 waves/block, block tile = 128 rows x 64 cols.
// B k-slab staged via async-to-LDS, double buffered; A software-pipelined;
// last iteration peeled so the steady-state loop is branch-free.
// ---------------------------------------------------------------------
__device__ inline void gemm_core(const bf16_t* A, const bf16_t* Bt,
                                 int Mrows, int Kdim, int row0, int col0,
                                 int tid, int lane, f32x8 (&acc)[4]) {
  __shared__ __align__(16) bf16_t Bs[2][64 * BSTRIDE];
  int arow = row0 + (lane & 15);
  if (arow >= Mrows) arow = Mrows - 1;
  const int nk = Kdim / 32;
  stage_b_async(Bt, Kdim, col0, 0, &Bs[0][0], tid);
  bf16x16 a_cur = load_a_frag(A, Kdim, arow, 0, lane);

  // steady state: iterations 0 .. nk-2 (branch-free body)
  for (int i = 0; i < nk - 1; ++i) {
    const int buf = i & 1;
    stage_b_async(Bt, Kdim, col0, (i + 1) * 32, &Bs[buf ^ 1][0], tid);
    wait_async1();                     // previous slab landed
    __syncthreads();
    __builtin_prefetch(A + (size_t)arow * Kdim + (i + 2) * 32, 0, 1);
    bf16x16 a_next = load_a_frag(A, Kdim, arow, (i + 1) * 32, lane);
#pragma unroll
    for (int t = 0; t < 4; ++t) {
      bf16x16 b = load_b_frag_lds(&Bs[buf][0], t * 16 + (lane & 15), lane);
      acc[t] = wmma_bf16(a_cur, b, acc[t]);
    }
    a_cur = a_next;
    __syncthreads();
  }

  // peeled final iteration
  wait_async0();
  __syncthreads();
  {
    const int buf = (nk - 1) & 1;
#pragma unroll
    for (int t = 0; t < 4; ++t) {
      bf16x16 b = load_b_frag_lds(&Bs[buf][0], t * 16 + (lane & 15), lane);
      acc[t] = wmma_bf16(a_cur, b, acc[t]);
    }
  }
}

// ---------------- small utility kernels ----------------
__global__ void f2bf_kernel(const float* in, bf16_t* out, int n) {
  int i = blockIdx.x * blockDim.x + threadIdx.x;
  int stride = gridDim.x * blockDim.x;
  for (; i < n; i += stride) out[i] = (bf16_t)in[i];
}

__global__ void ln_kernel(const float* xin, const float* w, const float* bb,
                          bf16_t* out, int ntok) {
  int t = blockIdx.x;
  int tid = threadIdx.x;
  if (t >= ntok) return;
  __shared__ float red[256];
  const float* xr = xin + (size_t)t * Cdim;
  float s = 0.f;
  for (int c = tid; c < Cdim; c += 256) s += xr[c];
  __syncthreads(); red[tid] = s; __syncthreads();
  for (int k = 128; k > 0; k >>= 1) { if (tid < k) red[tid] += red[tid + k]; __syncthreads(); }
  float mu = red[0] / Cdim;
  float vs = 0.f;
  for (int c = tid; c < Cdim; c += 256) { float d = xr[c] - mu; vs += d * d; }
  __syncthreads(); red[tid] = vs; __syncthreads();
  for (int k = 128; k > 0; k >>= 1) { if (tid < k) red[tid] += red[tid + k]; __syncthreads(); }
  float rinv = rsqrtf(red[0] / Cdim + LN_EPS);
  for (int c = tid; c < Cdim; c += 256)
    out[(size_t)t * Cdim + c] = (bf16_t)((xr[c] - mu) * rinv * w[c] + bb[c]);
}

// ---------------- WMMA GEMM kernels ----------------
__global__ __launch_bounds__(256)
void gemm_qkv_kernel(const bf16_t* A, const bf16_t* Bt, const float* bias,
                     bf16_t* qo, bf16_t* ko, bf16_t* vTo, int Mrows, int Kdim) {
  const int tid  = threadIdx.x;
  const int lane = tid & 31;
  const int wv   = tid >> 5;
  const int row0 = blockIdx.x * 128 + wv * 16;
  const int col0 = blockIdx.y * 64;
  f32x8 acc[4] = {};
  gemm_core(A, Bt, Mrows, Kdim, row0, col0, tid, lane, acc);

  const int g  = blockIdx.y;      // p = g/12, head = g%12
  const int p  = g / Hn;
  const int hh = g % Hn;
  const int rr = (lane & 16) ? 8 : 0;
  const int cc = lane & 15;
#pragma unroll
  for (int t = 0; t < 4; ++t) {
#pragma unroll
    for (int r = 0; r < 8; ++r) {
      int row = row0 + rr + r;
      if (row < Mrows) {
        int dh = t * 16 + cc;
        float v = acc[t][r] + bias[col0 + dh];
        int bbatch = row / Nseq;
        int n = row % Nseq;
        if (p == 0)
          qo[((size_t)(bbatch * Hn + hh) * Nseq + n) * DHd + dh] = (bf16_t)v;
        else if (p == 1)
          ko[((size_t)(bbatch * Hn + hh) * Nseq + n) * DHd + dh] = (bf16_t)v;
        else
          vTo[((size_t)(bbatch * Hn + hh) * DHd + dh) * NPAD + n] = (bf16_t)v;
      }
    }
  }
}

__global__ __launch_bounds__(256)
void gemm_res_f32_kernel(const bf16_t* A, const bf16_t* Bt, const float* bias,
                         const float* res, float* out, int Mrows, int Kdim, int Ncols) {
  const int tid  = threadIdx.x;
  const int lane = tid & 31;
  const int wv   = tid >> 5;
  const int row0 = blockIdx.x * 128 + wv * 16;
  const int col0 = blockIdx.y * 64;
  f32x8 acc[4] = {};
  gemm_core(A, Bt, Mrows, Kdim, row0, col0, tid, lane, acc);

  const int rr = (lane & 16) ? 8 : 0;
  const int cc = lane & 15;
#pragma unroll
  for (int t = 0; t < 4; ++t) {
#pragma unroll
    for (int r = 0; r < 8; ++r) {
      int row = row0 + rr + r;
      int col = col0 + t * 16 + cc;
      if (row < Mrows)
        out[(size_t)row * Ncols + col] =
            res[(size_t)row * Ncols + col] + bias[col] + acc[t][r];
    }
  }
}

__global__ __launch_bounds__(256)
void gemm_gelu_bf16_kernel(const bf16_t* A, const bf16_t* Bt, const float* bias,
                           bf16_t* out, int Mrows, int Kdim, int Ncols) {
  const int tid  = threadIdx.x;
  const int lane = tid & 31;
  const int wv   = tid >> 5;
  const int row0 = blockIdx.x * 128 + wv * 16;
  const int col0 = blockIdx.y * 64;
  f32x8 acc[4] = {};
  gemm_core(A, Bt, Mrows, Kdim, row0, col0, tid, lane, acc);

  const int rr = (lane & 16) ? 8 : 0;
  const int cc = lane & 15;
#pragma unroll
  for (int t = 0; t < 4; ++t) {
#pragma unroll
    for (int r = 0; r < 8; ++r) {
      int row = row0 + rr + r;
      int col = col0 + t * 16 + cc;
      if (row < Mrows) {
        float v = acc[t][r] + bias[col];
        float gelu = 0.5f * v * (1.0f + erff(v * 0.7071067811865475f));
        out[(size_t)row * Ncols + col] = (bf16_t)gelu;
      }
    }
  }
}

// ---------------- flash attention (one wave per 16-query tile) ----------------
__global__ __launch_bounds__(32)
void attn_kernel(const bf16_t* qm, const bf16_t* km, const bf16_t* vTm, bf16_t* om) {
  const int lane = threadIdx.x;
  const int qt = blockIdx.x, h = blockIdx.y, b = blockIdx.z;
  const int qbase = qt * 16;

  __shared__ float Smat[16][33];
  __shared__ __align__(16) bf16_t Pmat[16][32];
  __shared__ float rowf[16];

  const bf16_t* qp = qm + (size_t)(b * Hn + h) * Nseq * DHd;
  const bf16_t* kp = km + (size_t)(b * Hn + h) * Nseq * DHd;
  const bf16_t* vp = vTm + (size_t)(b * Hn + h) * DHd * NPAD;

  int qr = qbase + (lane & 15); if (qr >= Nseq) qr = Nseq - 1;
  const bf16x16 aq0 = load_a_frag(qp, DHd, qr, 0, lane);
  const bf16x16 aq1 = load_a_frag(qp, DHd, qr, 32, lane);

  f32x8 o0 = {}, o1 = {}, o2 = {}, o3 = {};
  float m_i = -3.0e38f, l_i = 0.f;      // valid in lanes < 16 (row = lane)

  for (int j0 = 0; j0 < Nseq; j0 += 32) {
    f32x8 s0 = {}, s1 = {};
    {
      int key0 = j0 + (lane & 15);      if (key0 >= Nseq) key0 = Nseq - 1;
      int key1 = j0 + 16 + (lane & 15); if (key1 >= Nseq) key1 = Nseq - 1;
      bf16x16 bk;
      bk = load_b_frag(kp, DHd, key0, 0, lane);  s0 = wmma_bf16(aq0, bk, s0);
      bk = load_b_frag(kp, DHd, key0, 32, lane); s0 = wmma_bf16(aq1, bk, s0);
      bk = load_b_frag(kp, DHd, key1, 0, lane);  s1 = wmma_bf16(aq0, bk, s1);
      bk = load_b_frag(kp, DHd, key1, 32, lane); s1 = wmma_bf16(aq1, bk, s1);
    }
    {
      const int rr = (lane & 16) ? 8 : 0;
      const int cc = lane & 15;
      const bool v0 = (j0 + cc) < Nseq;
      const bool v1 = (j0 + 16 + cc) < Nseq;
#pragma unroll
      for (int r = 0; r < 8; ++r) {
        Smat[rr + r][cc]      = v0 ? s0[r] * SCALE_F : -3.0e38f;
        Smat[rr + r][16 + cc] = v1 ? s1[r] * SCALE_F : -3.0e38f;
      }
    }
    __syncthreads();
    if (lane < 16) {
      float mx = m_i;
#pragma unroll
      for (int c = 0; c < 32; ++c) mx = fmaxf(mx, Smat[lane][c]);
      float sc = __expf(m_i - mx);
      float sum = 0.f;
#pragma unroll
      for (int c = 0; c < 32; ++c) {
        float p = __expf(Smat[lane][c] - mx);
        Pmat[lane][c] = (bf16_t)p;
        sum += p;
      }
      l_i = l_i * sc + sum;
      m_i = mx;
      rowf[lane] = sc;
    }
    __syncthreads();
    {
      const int rr = (lane & 16) ? 8 : 0;
#pragma unroll
      for (int r = 0; r < 8; ++r) {
        float f = rowf[rr + r];
        o0[r] *= f; o1[r] *= f; o2[r] *= f; o3[r] *= f;
      }
    }
    bf16x16 ap = load_a_frag(&Pmat[0][0], 32, lane & 15, 0, lane);
    bf16x16 bv;
    bv = load_b_frag(vp, NPAD,  0 + (lane & 15), j0, lane); o0 = wmma_bf16(ap, bv, o0);
    bv = load_b_frag(vp, NPAD, 16 + (lane & 15), j0, lane); o1 = wmma_bf16(ap, bv, o1);
    bv = load_b_frag(vp, NPAD, 32 + (lane & 15), j0, lane); o2 = wmma_bf16(ap, bv, o2);
    bv = load_b_frag(vp, NPAD, 48 + (lane & 15), j0, lane); o3 = wmma_bf16(ap, bv, o3);
    __syncthreads();
  }
  if (lane < 16) rowf[lane] = 1.0f / l_i;
  __syncthreads();
  const int rr = (lane & 16) ? 8 : 0;
  const int cc = lane & 15;
#pragma unroll
  for (int r = 0; r < 8; ++r) {
    int n = qbase + rr + r;
    if (n < Nseq) {
      float inv = rowf[rr + r];
      size_t base = ((size_t)(b * Nseq + n)) * Cdim + h * DHd;
      om[base + cc]      = (bf16_t)(o0[r] * inv);
      om[base + 16 + cc] = (bf16_t)(o1[r] * inv);
      om[base + 32 + cc] = (bf16_t)(o2[r] * inv);
      om[base + 48 + cc] = (bf16_t)(o3[r] * inv);
    }
  }
}

// ---------------- cls/priv attention rows (full softmax, rows 0 & 1) ----------
__global__ void cls_attn_kernel(const bf16_t* qm, const bf16_t* km,
                                float* act, float* prv) {
  const int b = blockIdx.x, tid = threadIdx.x;
  __shared__ float red[256];
  __shared__ float qrow[DHd];
  constexpr int U = (Nseq + 255) / 256;   // 7
  float accA[U], accP[U];
#pragma unroll
  for (int u = 0; u < U; ++u) { accA[u] = 0.f; accP[u] = 0.f; }
  for (int h = 0; h < Hn; ++h) {
    for (int row = 0; row < 2; ++row) {
      const bf16_t* qp = qm + ((size_t)(b * Hn + h) * Nseq + row) * DHd;
      if (tid < DHd) qrow[tid] = (float)qp[tid];
      __syncthreads();
      float lg[U];
      float mx = -3.0e38f;
      for (int u = 0; u < U; ++u) {
        int j = tid + u * 256;
        float d = -3.0e38f;
        if (j < Nseq) {
          const bf16_t* kpp = km + ((size_t)(b * Hn + h) * Nseq + j) * DHd;
          float s = 0.f;
          for (int c = 0; c < DHd; ++c) s += qrow[c] * (float)kpp[c];
          d = s * SCALE_F;
        }
        lg[u] = d; mx = fmaxf(mx, d);
      }
      __syncthreads(); red[tid] = mx; __syncthreads();
      for (int k = 128; k > 0; k >>= 1) { if (tid < k) red[tid] = fmaxf(red[tid], red[tid + k]); __syncthreads(); }
      float gmax = red[0];
      float ls = 0.f;
      for (int u = 0; u < U; ++u) {
        int j = tid + u * 256;
        if (j < Nseq) { float e = __expf(lg[u] - gmax); lg[u] = e; ls += e; }
        else lg[u] = 0.f;
      }
      __syncthreads(); red[tid] = ls; __syncthreads();
      for (int k = 128; k > 0; k >>= 1) { if (tid < k) red[tid] += red[tid + k]; __syncthreads(); }
      float inv = 1.0f / (red[0] * (float)Hn);   // includes 1/H head mean
      for (int u = 0; u < U; ++u) {
        int j = tid + u * 256;
        if (j >= 2 && j < Nseq) {
          float p = lg[u] * inv;
          if (row == 0) accA[u] += p; else accP[u] += p;
        }
      }
      __syncthreads();
    }
  }
  for (int u = 0; u < U; ++u) {
    int j = tid + u * 256;
    if (j >= 2 && j < Nseq) {
      act[b * Mtok + (j - 2)] = accA[u];
      prv[b * Mtok + (j - 2)] = accP[u];
    }
  }
}

__global__ void score_kernel(float* act, float* prv, float* score) {
  const int b = blockIdx.x, tid = threadIdx.x;
  __shared__ float red[256];
  float sA = 0.f, sP = 0.f;
  for (int m = tid; m < Mtok; m += 256) { sA += act[b * Mtok + m]; sP += prv[b * Mtok + m]; }
  __syncthreads(); red[tid] = sA; __syncthreads();
  for (int k = 128; k > 0; k >>= 1) { if (tid < k) red[tid] += red[tid + k]; __syncthreads(); }
  float gA = red[0];
  __syncthreads(); red[tid] = sP; __syncthreads();
  for (int k = 128; k > 0; k >>= 1) { if (tid < k) red[tid] += red[tid + k]; __syncthreads(); }
  float gP = red[0];
  for (int m = tid; m < Mtok; m += 256) {
    float a = act[b * Mtok + m] / gA;
    float p = prv[b * Mtok + m] / gP;
    act[b * Mtok + m] = a;                 // normalized act kept for extra token
    score[b * Mtok + m] = a - 0.5f * p;
  }
}

// top-K by rank counting (descending, index tie-break), then stable-sort complement
__global__ void topk_kernel(const float* score, int* sel, int* cmpl) {
  const int b = blockIdx.x, tid = threadIdx.x;
  __shared__ float sc[Mtok];
  __shared__ int   msk[Mtok];
  for (int i = tid; i < Mtok; i += 256) sc[i] = score[b * Mtok + i];
  __syncthreads();
  for (int i = tid; i < Mtok; i += 256) {
    float si = sc[i];
    int r = 0;
    for (int j = 0; j < Mtok; ++j) {
      float sj = sc[j];
      r += (sj > si) || (sj == si && j < i);
    }
    if (r < KSEL) sel[b * KSEL + r] = i;
    msk[i] = (r < KSEL) ? 0 : i;           // reference: masked[idx]=0
  }
  __syncthreads();
  for (int i = tid; i < Mtok; i += 256) {
    int vi = msk[i];
    int r = 0;
    for (int j = 0; j < Mtok; ++j) {
      int vj = msk[j];
      r += (vj < vi) || (vj == vi && j < i);
    }
    if (r >= KSEL) cmpl[b * CKn + (r - KSEL)] = vi;
  }
}

__global__ void build_x2_kernel(const float* x1, const int* sel, const int* cmpl,
                                const float* actn, float* x2) {
  const int r = blockIdx.x, b = blockIdx.y, tid = threadIdx.x;
  float* dst = x2 + ((size_t)(b * N2 + r)) * Cdim;
  if (r < 2 + KSEL) {
    int srow = (r < 2) ? r : (2 + sel[b * KSEL + (r - 2)]);
    const float* src = x1 + ((size_t)(b * Nseq + srow)) * Cdim;
    for (int c = tid; c < Cdim; c += 256) dst[c] = src[c];
  } else {
    float acc[3] = {0.f, 0.f, 0.f};
    for (int t = 0; t < CKn; ++t) {
      int id = cmpl[b * CKn + t];
      float w = actn[b * Mtok + id];
      const float* s = x1 + ((size_t)(b * Nseq + 2 + id)) * Cdim;
#pragma unroll
      for (int u = 0; u < 3; ++u) {
        int c = tid + u * 256;
        acc[u] += s[c] * w;
      }
    }
#pragma unroll
    for (int u = 0; u < 3; ++u) dst[tid + u * 256] = acc[u];
  }
}

// ---------------- host launcher ----------------
extern "C" void kernel_launch(void* const* d_in, const int* in_sizes, int n_in,
                              void* d_out, int out_size, void* d_ws, size_t ws_size,
                              hipStream_t stream) {
  const float* x      = (const float*)d_in[0];
  const float* ln1_w  = (const float*)d_in[1];
  const float* ln1_b  = (const float*)d_in[2];
  const float* qkv_w  = (const float*)d_in[3];
  const float* qkv_b  = (const float*)d_in[4];
  const float* proj_w = (const float*)d_in[5];
  const float* proj_b = (const float*)d_in[6];
  const float* ln2_w  = (const float*)d_in[7];
  const float* ln2_b  = (const float*)d_in[8];
  const float* fc1_w  = (const float*)d_in[9];
  const float* fc1_b  = (const float*)d_in[10];
  const float* fc2_w  = (const float*)d_in[11];
  const float* fc2_b  = (const float*)d_in[12];
  float* outp = (float*)d_out;

  char* ws = (char*)d_ws;
  size_t off = 0;
  auto alloc = [&](size_t bytes) -> void* {
    void* p = ws + off;
    off = (off + bytes + 255) & ~(size_t)255;
    return p;
  };
  bf16_t* h_bf  = (bf16_t*)alloc((size_t)TOK1 * Cdim * 2);
  bf16_t* q_bf  = (bf16_t*)alloc((size_t)Bq * Hn * Nseq * DHd * 2);
  bf16_t* k_bf  = (bf16_t*)alloc((size_t)Bq * Hn * Nseq * DHd * 2);
  bf16_t* vT_bf = (bf16_t*)alloc((size_t)Bq * Hn * DHd * NPAD * 2);
  bf16_t* o_bf  = (bf16_t*)alloc((size_t)TOK1 * Cdim * 2);
  float*  x1    = (float*) alloc((size_t)TOK1 * Cdim * 4);
  float*  act   = (float*) alloc((size_t)Bq * Mtok * 4);
  float*  prv   = (float*) alloc((size_t)Bq * Mtok * 4);
  float*  scre  = (float*) alloc((size_t)Bq * Mtok * 4);
  int*    sel   = (int*)   alloc((size_t)Bq * KSEL * 4);
  int*    cmpl  = (int*)   alloc((size_t)Bq * CKn * 4);
  float*  x2    = (float*) alloc((size_t)TOK2 * Cdim * 4);
  bf16_t* h2_bf = (bf16_t*)alloc((size_t)TOK2 * Cdim * 2);
  bf16_t* g_bf  = (bf16_t*)alloc((size_t)TOK2 * 4 * Cdim * 2);
  bf16_t* wq_bf = (bf16_t*)alloc((size_t)3 * Cdim * Cdim * 2);
  bf16_t* wp_bf = (bf16_t*)alloc((size_t)Cdim * Cdim * 2);
  bf16_t* w1_bf = (bf16_t*)alloc((size_t)4 * Cdim * Cdim * 2);
  bf16_t* w2_bf = (bf16_t*)alloc((size_t)Cdim * 4 * Cdim * 2);

  // weight conversions fp32 -> bf16 (layouts already "B column-major")
  f2bf_kernel<<<2048, 256, 0, stream>>>(qkv_w, wq_bf, 3 * Cdim * Cdim);
  f2bf_kernel<<<2048, 256, 0, stream>>>(proj_w, wp_bf, Cdim * Cdim);
  f2bf_kernel<<<2048, 256, 0, stream>>>(fc1_w, w1_bf, 4 * Cdim * Cdim);
  f2bf_kernel<<<2048, 256, 0, stream>>>(fc2_w, w2_bf, Cdim * 4 * Cdim);

  // zero V^T padding region (whole buffer) so masked-key reads are clean
  (void)hipMemsetAsync(vT_bf, 0, (size_t)Bq * Hn * DHd * NPAD * 2, stream);

  // LN1
  ln_kernel<<<TOK1, 256, 0, stream>>>(x, ln1_w, ln1_b, h_bf, TOK1);

  // QKV GEMM: [6280,768] x [768,2304]
  {
    dim3 grid((TOK1 + 127) / 128, (3 * Cdim) / 64);
    gemm_qkv_kernel<<<grid, 256, 0, stream>>>(h_bf, wq_bf, qkv_b,
                                              q_bf, k_bf, vT_bf, TOK1, Cdim);
  }

  // attention
  {
    dim3 grid((Nseq + 15) / 16, Hn, Bq);
    attn_kernel<<<grid, 32, 0, stream>>>(q_bf, k_bf, vT_bf, o_bf);
  }

  // proj + residual -> x1
  {
    dim3 grid((TOK1 + 127) / 128, Cdim / 64);
    gemm_res_f32_kernel<<<grid, 256, 0, stream>>>(o_bf, wp_bf, proj_b, x, x1,
                                                  TOK1, Cdim, Cdim);
  }

  // token scoring + selection
  cls_attn_kernel<<<Bq, 256, 0, stream>>>(q_bf, k_bf, act, prv);
  score_kernel<<<Bq, 256, 0, stream>>>(act, prv, scre);
  topk_kernel<<<Bq, 256, 0, stream>>>(scre, sel, cmpl);
  {
    dim3 grid(N2, Bq);
    build_x2_kernel<<<grid, 256, 0, stream>>>(x1, sel, cmpl, act, x2);
  }

  // LN2 + MLP
  ln_kernel<<<TOK2, 256, 0, stream>>>(x2, ln2_w, ln2_b, h2_bf, TOK2);
  {
    dim3 grid((TOK2 + 127) / 128, (4 * Cdim) / 64);
    gemm_gelu_bf16_kernel<<<grid, 256, 0, stream>>>(h2_bf, w1_bf, fc1_b, g_bf,
                                                    TOK2, Cdim, 4 * Cdim);
  }
  {
    dim3 grid((TOK2 + 127) / 128, Cdim / 64);
    gemm_res_f32_kernel<<<grid, 256, 0, stream>>>(g_bf, w2_bf, fc2_b, x2, outp,
                                                  TOK2, 4 * Cdim, Cdim);
  }
  (void)in_sizes; (void)n_in; (void)out_size; (void)ws_size;
}
